// GroupMat_66838281060838
// MI455X (gfx1250) — matrix operand order repaired and checked by reading, not compiled, not run on hardware
//
#include <hip/hip_runtime.h>
#include <hip/hip_bf16.h>

#define NN    800
#define NF    111
#define EMBD  32
#define HD    200
#define NCLS  15
#define NE_   8000
#define ALPHA_ 0.5f
#define LNEPS  1e-5f

typedef __attribute__((ext_vector_type(16))) _Float16 v16h;
typedef __attribute__((ext_vector_type(8)))  _Float16 v8h;
typedef __attribute__((ext_vector_type(8)))  float    v8f;

// =====================================================================
// WMMA GEMM:  C[M,N] = beta*C + alpha*(*aptr)*(op(A) @ B) + gamma*I + bscale*bias[n]
// op(A)=A ([M,K], lda=K) or A^T (physical [K,M], lda=M) when transA!=0.
// Block: 128 threads = 4 wave32; tile 64x64; BK=32; double-buffered LDS.
// Wave w owns 16-row strip m=w*16 and all 64 cols: 4 WMMA accs, A-frag reuse.
// LDS fragment-contiguous layouts -> each fragment is ONE v16h (2x ds_load_b128):
//   As2[m][pos(k)], pos = [k0..7, k16..23, k8..15, k24..31]
//   Bs2[n][pos(k)], same pos as As2
// All 5 fragment loads are issued before the WMMA chain so the scheduler can
// use partial s_wait_dscnt instead of draining LDS before every WMMA.
// =====================================================================
__global__ __launch_bounds__(128) void wgemm(
    const float* __restrict__ A, const float* __restrict__ B, float* __restrict__ C,
    int M, int N, int K, int lda, int ldb, int ldc, int transA,
    float alpha, const float* __restrict__ aptr, float beta, float gamma,
    const float* __restrict__ bias, float bscale)
{
  __shared__ alignas(128) _Float16 As2[2][64][48];  // 96B row stride (32B aligned)
  __shared__ alignas(128) _Float16 Bs2[2][64][48];
  const int tid  = threadIdx.x;
  const int lane = tid & 31;
  const int wm   = tid >> 5;        // wave id 0..3 -> 16-row m strip
  const int hi   = lane >> 4;
  const int l16  = lane & 15;
  const int bm0  = blockIdx.y * 64;
  const int bn0  = blockIdx.x * 64;
  float al = alpha; if (aptr) al *= aptr[0];
  v8f acc[4] = {{}, {}, {}, {}};

  const int sm = tid >> 1;          // 0..63: A row / B col owned for staging
  const int kg = (tid & 1) * 16;    // 0 or 16: k-half of the stage owned
  const int gm  = bm0 + sm;
  const int gn  = bn0 + sm;
  const bool mok = gm < M;
  const bool nok = gn < N;
  const int gmc = mok ? gm : (M - 1);
  const int gnc = nok ? gn : (N - 1);
  const bool fastA = (!transA) && ((lda & 3) == 0);

  // ---- branch-free stage loads (issue all loads, select after) ----
  auto loadA = [&](int k0, v8h& c0, v8h& c1) {
    if (fastA && mok && (k0 + 32 <= K)) {
      const float* base = &A[(long)gm * lda + k0 + kg];   // 16B aligned
      float4 f0 = *(const float4*)(base + 0);
      float4 f1 = *(const float4*)(base + 4);
      float4 f2 = *(const float4*)(base + 8);
      float4 f3 = *(const float4*)(base + 12);
      c0[0]=(_Float16)f0.x; c0[1]=(_Float16)f0.y; c0[2]=(_Float16)f0.z; c0[3]=(_Float16)f0.w;
      c0[4]=(_Float16)f1.x; c0[5]=(_Float16)f1.y; c0[6]=(_Float16)f1.z; c0[7]=(_Float16)f1.w;
      c1[0]=(_Float16)f2.x; c1[1]=(_Float16)f2.y; c1[2]=(_Float16)f2.z; c1[3]=(_Float16)f2.w;
      c1[4]=(_Float16)f3.x; c1[5]=(_Float16)f3.y; c1[6]=(_Float16)f3.z; c1[7]=(_Float16)f3.w;
    } else {
#pragma unroll
      for (int j = 0; j < 8; ++j) {
        int ga = k0 + kg + j, gb = ga + 8;
        int gac = ga < K ? ga : (K - 1);
        int gbc = gb < K ? gb : (K - 1);
        float va = transA ? A[(long)gac * lda + gmc] : A[(long)gmc * lda + gac];
        float vb = transA ? A[(long)gbc * lda + gmc] : A[(long)gmc * lda + gbc];
        c0[j] = (mok && ga < K) ? (_Float16)va : (_Float16)0.f;
        c1[j] = (mok && gb < K) ? (_Float16)vb : (_Float16)0.f;
      }
    }
  };
  auto loadB = [&](int k0, v8h& c0, v8h& c1) {
#pragma unroll
    for (int j = 0; j < 8; ++j) {
      int ga = k0 + kg + j, gb = ga + 8;
      int gac = ga < K ? ga : (K - 1);
      int gbc = gb < K ? gb : (K - 1);
      float va = B[(long)gac * ldb + gnc];
      float vb = B[(long)gbc * ldb + gnc];
      c0[j] = (nok && ga < K) ? (_Float16)va : (_Float16)0.f;
      c1[j] = (nok && gb < K) ? (_Float16)vb : (_Float16)0.f;
    }
  };
  const int pA = (kg == 0) ? 0 : 8;   // A perm base for chunk [kg, kg+8)
  auto storeA = [&](int buf, v8h c0, v8h c1) {
    *(v8h*)&As2[buf][sm][pA]      = c0;
    *(v8h*)&As2[buf][sm][pA + 16] = c1;   // chunk [kg+8, kg+16)
  };
  auto storeB = [&](int buf, v8h c0, v8h c1) {
    *(v8h*)&Bs2[buf][sm][pA]      = c0;
    *(v8h*)&Bs2[buf][sm][pA + 16] = c1;
  };

  const int nS = (K + 31) >> 5;
  v8h a0, a1, b0, b1;
  loadA(0, a0, a1); loadB(0, b0, b1);
  storeA(0, a0, a1); storeB(0, b0, b1);

  for (int s = 0; s < nS; ++s) {
    __syncthreads();                       // lds[s&1] ready for all waves
    const int k0n = (s + 1) << 5;
    if (k0n < K) {                         // issue next-stage global loads now
      loadA(k0n, a0, a1);
      loadB(k0n, b0, b1);
      if (k0n + 32 < K) {                  // -> global_prefetch_b8
        __builtin_prefetch(&B[(long)(k0n + 32) * ldb + bn0]);
        __builtin_prefetch(transA ? &A[(long)(k0n + 32) * lda + bm0]
                                  : &A[(long)bm0 * lda + (k0n + 32)]);
      }
    }
    const int buf = s & 1;
    // issue ALL fragment loads first (in-order DS returns -> partial waits)
    v16h af  = *(const v16h*)&As2[buf][wm * 16 + l16][hi * 16];
    v16h bf0 = *(const v16h*)&Bs2[buf][ 0 + l16][hi * 16];
    v16h bf1 = *(const v16h*)&Bs2[buf][16 + l16][hi * 16];
    v16h bf2 = *(const v16h*)&Bs2[buf][32 + l16][hi * 16];
    v16h bf3 = *(const v16h*)&Bs2[buf][48 + l16][hi * 16];
    acc[0] = __builtin_amdgcn_wmma_f32_16x16x32_f16(false, af, false, bf0,
                                                    (short)0, acc[0], false, false);
    acc[1] = __builtin_amdgcn_wmma_f32_16x16x32_f16(false, af, false, bf1,
                                                    (short)0, acc[1], false, false);
    acc[2] = __builtin_amdgcn_wmma_f32_16x16x32_f16(false, af, false, bf2,
                                                    (short)0, acc[2], false, false);
    acc[3] = __builtin_amdgcn_wmma_f32_16x16x32_f16(false, af, false, bf3,
                                                    (short)0, acc[3], false, false);
    if (k0n < K) {                         // fill the alternate buffer
      storeA(buf ^ 1, a0, a1);
      storeB(buf ^ 1, b0, b1);
    }
  }
  // epilogue: C/D layout -> row = v + 8*hi, col = l16 (+16*t)
#pragma unroll
  for (int v = 0; v < 8; ++v) {
    int m = bm0 + wm * 16 + v + hi * 8;
    if (m >= M) continue;
#pragma unroll
    for (int t = 0; t < 4; ++t) {
      int n = bn0 + t * 16 + l16;
      if (n < N) {
        long idx = (long)m * ldc + n;
        float r = al * acc[t][v];
        if (beta != 0.f) r += beta * C[idx];
        if (gamma != 0.f && m == n) r += gamma;
        if (bias) r += bscale * bias[n];
        C[idx] = r;
      }
    }
  }
}

// ================= element-wise / reduction kernels =================
__global__ void k_fill(float* p, float v, long n) {
  long i = (long)blockIdx.x * blockDim.x + threadIdx.x;
  if (i < n) p[i] = v;
}
__global__ void k_diag(float* p, float d, int N) {
  long i = (long)blockIdx.x * blockDim.x + threadIdx.x;
  if (i < (long)N * N) { int r = (int)(i / N), c = (int)(i % N); p[i] = (r == c) ? d : 0.f; }
}
__global__ void k_papp(float* papp, const float* tri, const float* invn, long n, int N) {
  long i = (long)blockIdx.x * blockDim.x + threadIdx.x;
  if (i < n) { int r = (int)(i / N), c = (int)(i % N); papp[i] = ((r == c) ? 1.f : 0.f) - tri[i] * invn[0]; }
}
__global__ void k_paccum(float* ps, float* qs, const float* ph, float P, float Q, long n) {
  long i = (long)blockIdx.x * blockDim.x + threadIdx.x;
  if (i < n) { float v = ph[i]; ps[i] += P * v; qs[i] += Q * v; }
}
__global__ void k_tscale(float* X, const float* q, const float* s0, int N) {
  long i = (long)blockIdx.x * blockDim.x + threadIdx.x;
  if (i < (long)N * N) { int r = (int)(i / N), c = (int)(i % N); X[i] = q[(long)c * N + r] * s0[0]; }
}
__global__ void k_thresh(float* dst, const float* src, long n) {
  long i = (long)blockIdx.x * blockDim.x + threadIdx.x;
  if (i < n) dst[i] = (src[i] != 0.f) ? 1.f : 0.f;
}
__global__ void k_adj(float* A, const int* __restrict__ ei, int ne, int N) {
  int e = blockIdx.x * blockDim.x + threadIdx.x;
  if (e < ne) A[(long)ei[e] * N + ei[ne + e]] = 1.f;
}
__global__ void k_scatter(float* out, const float* __restrict__ x, const int* __restrict__ ei,
                          int grow, int srow, int ne, int F, float coef) {
  long idx = (long)blockIdx.x * blockDim.x + threadIdx.x;
  long e = idx / F; int f = (int)(idx % F);
  if (e < ne) {
    int g = ei[(long)grow * ne + e];
    int s = ei[(long)srow * ne + e];
    atomicAdd(&out[(long)s * F + f], coef * x[(long)g * F + f]);
  }
}
__global__ void k_rowbias(float* out, const float* b, float scale, long n, int F) {
  long i = (long)blockIdx.x * blockDim.x + threadIdx.x;
  if (i < n) out[i] += scale * b[i % F];
}
__global__ void k_meanvar(const float* __restrict__ x, long n, float* scal, float eps) {
  __shared__ float s1[256], s2[256];
  float a = 0.f, b = 0.f;
  for (long i = threadIdx.x; i < n; i += 256) { float v = x[i]; a += v; b += v * v; }
  s1[threadIdx.x] = a; s2[threadIdx.x] = b; __syncthreads();
  for (int s = 128; s > 0; s >>= 1) {
    if ((int)threadIdx.x < s) { s1[threadIdx.x] += s1[threadIdx.x + s]; s2[threadIdx.x] += s2[threadIdx.x + s]; }
    __syncthreads();
  }
  if (threadIdx.x == 0) {
    float mu = s1[0] / (float)n;
    float var = s2[0] / (float)n - mu * mu;
    scal[0] = mu; scal[1] = rsqrtf(var + eps);
  }
}
__global__ void k_ln(float* out, const float* in, const float* scal,
                     const float* w, const float* b, long n, int F) {
  long i = (long)blockIdx.x * blockDim.x + threadIdx.x;
  if (i < n) { int f = (int)(i % F); out[i] = (in[i] - scal[0]) * scal[1] * w[f] + b[f]; }
}
__global__ void k_frob(const float* __restrict__ x, long n, float* scal) {
  __shared__ float s1[256];
  float a = 0.f;
  for (long i = threadIdx.x; i < n; i += 256) { float v = x[i]; a += v * v; }
  s1[threadIdx.x] = a; __syncthreads();
  for (int s = 128; s > 0; s >>= 1) {
    if ((int)threadIdx.x < s) s1[threadIdx.x] += s1[threadIdx.x + s];
    __syncthreads();
  }
  if (threadIdx.x == 0) {
    float nA = sqrtf(s1[0]);
    scal[0] = nA; scal[1] = sqrtf(nA); scal[2] = 1.f / nA;
  }
}
__global__ void k_norms(const float* __restrict__ q, int N, float* out) {
  __shared__ float sm[256];
  int tid = threadIdx.x;
  float mx = 0.f;
  for (int r = tid; r < N; r += 256) {
    float s = 0.f;
    for (int c = 0; c < N; ++c) s += fabsf(q[(long)r * N + c]);
    mx = fmaxf(mx, s);
  }
  sm[tid] = mx; __syncthreads();
  for (int s = 128; s > 0; s >>= 1) { if (tid < s) sm[tid] = fmaxf(sm[tid], sm[tid + s]); __syncthreads(); }
  float rinf = sm[0];
  __syncthreads();
  mx = 0.f;
  for (int c = tid; c < N; c += 256) {
    float s = 0.f;
    for (int r = 0; r < N; ++r) s += fabsf(q[(long)r * N + c]);
    mx = fmaxf(mx, s);
  }
  sm[tid] = mx; __syncthreads();
  for (int s = 128; s > 0; s >>= 1) { if (tid < s) sm[tid] = fmaxf(sm[tid], sm[tid + s]); __syncthreads(); }
  if (tid == 0) out[0] = 1.f / (sm[0] * rinf);
}
// pairwise sigmoid((z_i - z_j)^2 . w + b) on upper triangle (incl. diag), 0 below
__global__ __launch_bounds__(256) void k_pairwise(
    float* __restrict__ tri, const float* __restrict__ z,
    const float* __restrict__ linW, const float* __restrict__ linb, int N) {
  __shared__ float zi[16][HD];
  __shared__ float zj[16][HD];
  __shared__ float wsh[HD];
  const int tid = threadIdx.x;
  const int i0 = blockIdx.y * 16;
  const int j0 = blockIdx.x * 16;
  const bool live = (j0 + 15) >= i0;
  if (live) {
    for (int idx = tid; idx < 16 * HD; idx += 256) {
      int r = idx / HD, f = idx % HD;
      int gi = i0 + r, gj = j0 + r;
      zi[r][f] = (gi < N) ? z[(long)gi * HD + f] : 0.f;
      zj[r][f] = (gj < N) ? z[(long)gj * HD + f] : 0.f;
    }
    for (int f = tid; f < HD; f += 256) wsh[f] = linW[f];
    __syncthreads();
  }
  int ti = tid >> 4, tj = tid & 15;
  int i = i0 + ti, j = j0 + tj;
  if (i < N && j < N) {
    float val = 0.f;
    if (live && j >= i) {
      float acc = linb[0];
      for (int f = 0; f < HD; ++f) {
        float d = zi[ti][f] - zj[tj][f];
        acc += d * d * wsh[f];
      }
      val = 1.f / (1.f + expf(-acc));
    }
    tri[(long)i * N + j] = val;
  }
}
__global__ void k_meanpool(const float* __restrict__ z, float* out, int N, int F) {
  for (int f = threadIdx.x; f < F; f += blockDim.x) {
    float s = 0.f;
    for (int r = 0; r < N; ++r) s += z[(long)r * F + f];
    out[f] = s / (float)N;
  }
}
__global__ void k_classifier(const float* __restrict__ x2, const float* __restrict__ W,
                             const float* __restrict__ b, float* out, int F, int C) {
  __shared__ float lg[32];
  __shared__ float lse;
  int t = threadIdx.x;
  if (t < C) {
    float s = b[t];
    for (int f = 0; f < F; ++f) s += x2[f] * W[f * C + t];
    lg[t] = s;
  }
  __syncthreads();
  if (t == 0) {
    float m = lg[0];
    for (int c = 1; c < C; ++c) m = fmaxf(m, lg[c]);
    float s = 0.f;
    for (int c = 0; c < C; ++c) s += expf(lg[c] - m);
    lse = m + logf(s);
  }
  __syncthreads();
  if (t < C) out[t] = lg[t] - lse;
}

// =====================================================================
extern "C" void kernel_launch(void* const* d_in, const int* in_sizes, int n_in,
                              void* d_out, int out_size, void* d_ws, size_t ws_size,
                              hipStream_t stream) {
  (void)in_sizes; (void)n_in; (void)out_size; (void)ws_size;
  const float* x_note  = (const float*)d_in[0];
  const float* embed_W = (const float*)d_in[1];
  const float* embed_b = (const float*)d_in[2];
  const float* g1l1W = (const float*)d_in[3];
  const float* g1l1b = (const float*)d_in[4];
  const float* g1l2W = (const float*)d_in[5];
  const float* g1l2b = (const float*)d_in[6];
  const float* g1ln  = (const float*)d_in[7];
  const float* g2l1W = (const float*)d_in[8];
  const float* g2l1b = (const float*)d_in[9];
  const float* g2l2W = (const float*)d_in[10];
  const float* g2l2b = (const float*)d_in[11];
  const float* g2ln  = (const float*)d_in[12];
  const float* linW  = (const float*)d_in[13];
  const float* linb  = (const float*)d_in[14];
  const float* clfW  = (const float*)d_in[15];
  const float* clfb  = (const float*)d_in[16];
  const int* ei[4] = {(const int*)d_in[17], (const int*)d_in[18],
                      (const int*)d_in[19], (const int*)d_in[20]};

  const long NN2 = (long)NN * NN;           // 640000
  float* ws   = (float*)d_ws;
  float* SCAL = ws;                         // [0]mu [1]rstd [2]normA [3]sqrt [4]inv [5]s0
  float* H0   = ws + 64;                    // 800x32
  float* H1   = H0 + NN * EMBD;             // 800x32
  float* ACC  = H1 + NN * EMBD;             // 800x200
  float* TMP  = ACC + NN * HD;              // 800x200
  float* Z    = TMP + NN * HD;              // 800x200
  float* MK[4];
  MK[0] = Z + NN * HD;
  for (int e = 1; e < 4; ++e) MK[e] = MK[e - 1] + NN2;
  float* Bb[6];
  Bb[0] = MK[3] + NN2;
  for (int i = 1; i < 6; ++i) Bb[i] = Bb[i - 1] + NN2;

  float* x2o = (float*)d_out;
  float* lgo = x2o + HD;
  float* S1  = lgo + NCLS;
  float* S2  = S1 + NN2;

  auto gemm = [&](float* C, const float* A, const float* Bm, int M, int N, int K,
                  int lda, int ldb, int ldc, int transA,
                  float alpha, const float* aptr, float beta, float gamma,
                  const float* bias, float bscale) {
    dim3 g((N + 63) / 64, (M + 63) / 64);
    wgemm<<<g, 128, 0, stream>>>(A, Bm, C, M, N, K, lda, ldb, ldc, transA,
                                 alpha, aptr, beta, gamma, bias, bscale);
  };
  auto fill = [&](float* p, float v, long n) {
    k_fill<<<(int)((n + 255) / 256), 256, 0, stream>>>(p, v, n);
  };
  auto layernorm = [&](float* out, const float* in, const float* w, const float* b) {
    long n = (long)NN * HD;
    k_meanvar<<<1, 256, 0, stream>>>(in, n, SCAL, LNEPS);
    k_ln<<<(int)((n + 255) / 256), 256, 0, stream>>>(out, in, SCAL, w, b, n, HD);
  };

  auto sparse_layer = [&](const float* h, int Fin, const float* W, const float* b) {
    long n = (long)NN * HD;
    fill(ACC, 0.f, n);
    long sc = (long)NE_ * HD;
    int scb = (int)((sc + 255) / 256);
    for (int e = 0; e < 4; ++e) {
      const float* W0 = W + (long)(0 * 4 + e) * Fin * HD;
      const float* W1 = W + (long)(1 * 4 + e) * Fin * HD;
      const float* W2 = W + (long)(2 * 4 + e) * Fin * HD;
      const float* b0 = b + (0 * 4 + e) * HD;
      const float* b1 = b + (1 * 4 + e) * HD;
      const float* b2 = b + (2 * 4 + e) * HD;
      gemm(TMP, h, W0, NN, HD, Fin, Fin, HD, HD, 0, 1.f, nullptr, 0.f, 0.f, nullptr, 0.f);
      k_scatter<<<scb, 256, 0, stream>>>(ACC, TMP, ei[e], 0, 1, NE_, HD, 1.f - ALPHA_);
      k_rowbias<<<(int)((n + 255) / 256), 256, 0, stream>>>(ACC, b0, 1.f - ALPHA_, n, HD);
      gemm(TMP, h, W1, NN, HD, Fin, Fin, HD, HD, 0, 1.f, nullptr, 0.f, 0.f, nullptr, 0.f);
      k_scatter<<<scb, 256, 0, stream>>>(ACC, TMP, ei[e], 1, 0, NE_, HD, ALPHA_);
      k_rowbias<<<(int)((n + 255) / 256), 256, 0, stream>>>(ACC, b1, ALPHA_, n, HD);
      gemm(ACC, h, W2, NN, HD, Fin, Fin, HD, HD, 0, 1.f, nullptr, 1.f, 0.f, b2, 1.f);
    }
  };

  auto dense_layer = [&](const float* h, int Fin, const float* W, const float* b) {
    fill(ACC, 0.f, (long)NN * HD);
    for (int e = 0; e < 4; ++e) {
      const float* W0 = W + (long)(0 * 4 + e) * Fin * HD;
      const float* W1 = W + (long)(1 * 4 + e) * Fin * HD;
      const float* W2 = W + (long)(2 * 4 + e) * Fin * HD;
      const float* b0 = b + (0 * 4 + e) * HD;
      const float* b1 = b + (1 * 4 + e) * HD;
      const float* b2 = b + (2 * 4 + e) * HD;
      gemm(TMP, h, W0, NN, HD, Fin, Fin, HD, HD, 0, 1.f, nullptr, 0.f, 0.f, nullptr, 0.f);
      gemm(ACC, MK[e], TMP, NN, HD, NN, NN, HD, HD, 1, 1.f - ALPHA_, nullptr, 1.f, 0.f, b0, 1.f - ALPHA_);
      gemm(TMP, h, W1, NN, HD, Fin, Fin, HD, HD, 0, 1.f, nullptr, 0.f, 0.f, nullptr, 0.f);
      gemm(ACC, MK[e], TMP, NN, HD, NN, NN, HD, HD, 0, ALPHA_, nullptr, 1.f, 0.f, b1, ALPHA_);
      gemm(ACC, h, W2, NN, HD, Fin, Fin, HD, HD, 0, 1.f, nullptr, 1.f, 0.f, b2, 1.f);
    }
  };

  // [5/5] Pade of sqrt(1-x):  P = (1024,-2816,2816,-1232,220,-11)/1024, Q = (1024,-2304,1792,-560,60,-1)/1024
  static const float PP[6] = {1.f, -2.75f, 2.75f, -1.203125f, 0.21484375f, -0.0107421875f};
  static const float QQ[6] = {1.f, -2.25f, 1.75f, -0.546875f, 0.05859375f, -0.0009765625f};

  auto pairwise_S = [&](const float* z, float* Sout) {
    int eb = (int)((NN2 + 255) / 256);
    k_pairwise<<<dim3(50, 50), 256, 0, stream>>>(Bb[0], z, linW, linb, NN);
    k_frob<<<1, 256, 0, stream>>>(Bb[0], NN2, SCAL + 2);
    k_papp<<<eb, 256, 0, stream>>>(Bb[1], Bb[0], SCAL + 4, NN2, NN);
    k_diag<<<eb, 256, 0, stream>>>(Bb[4], PP[0], NN);   // p_sqrt = P0*I
    k_diag<<<eb, 256, 0, stream>>>(Bb[5], QQ[0], NN);   // q_sqrt = Q0*I
    float* ph = Bb[1];                                  // p_hat = p_app
    for (int i = 0; i < 5; ++i) {
      k_paccum<<<eb, 256, 0, stream>>>(Bb[4], Bb[5], ph, PP[i + 1], QQ[i + 1], NN2);
      if (i < 4) {
        float* nxt = (ph == Bb[2]) ? Bb[3] : Bb[2];
        gemm(nxt, ph, Bb[1], NN, NN, NN, NN, NN, NN, 0, 1.f, nullptr, 0.f, 0.f, nullptr, 0.f);
        ph = nxt;
      }
    }
    // Newton-Schulz inverse of q_sqrt (Bb[5]); X0 = q^T/(|q|_1 |q|_inf)
    k_norms<<<1, 256, 0, stream>>>(Bb[5], NN, SCAL + 5);
    float* Xa = Bb[0]; float* Xb = Bb[1];
    k_tscale<<<eb, 256, 0, stream>>>(Xa, Bb[5], SCAL + 5, NN);
    for (int it = 0; it < 30; ++it) {
      gemm(Bb[2], Bb[5], Xa, NN, NN, NN, NN, NN, NN, 0, -1.f, nullptr, 0.f, 2.f, nullptr, 0.f); // 2I - qX
      gemm(Xb, Xa, Bb[2], NN, NN, NN, NN, NN, NN, 0, 1.f, nullptr, 0.f, 0.f, nullptr, 0.f);
      float* t = Xa; Xa = Xb; Xb = t;
    }
    // S = X @ p_sqrt * sqrt(normA)
    gemm(Sout, Xa, Bb[4], NN, NN, NN, NN, NN, NN, 0, 1.f, SCAL + 3, 0.f, 0.f, nullptr, 0.f);
  };

  int eb = (int)((NN2 + 255) / 256);

  // 1) embedding: h0 = x_note @ embed_W + b
  gemm(H0, x_note, embed_W, NN, EMBD, NF, NF, EMBD, EMBD, 0, 1.f, nullptr, 0.f, 0.f, embed_b, 1.f);

  // 2) hetero_gnn #1 (sparse, g1 weights) -> z0 in Z
  sparse_layer(H0, EMBD, g1l1W, g1l1b);
  layernorm(Z, ACC, g1ln + 0, g1ln + HD);
  sparse_layer(Z, HD, g1l2W, g1l2b);
  layernorm(Z, ACC, g1ln + 2 * HD, g1ln + 3 * HD);

  // 3) S1 = mpa_lya(pairwise(z0))
  pairwise_S(Z, S1);

  // 4) h = S1 @ h0 -> H1
  gemm(H1, S1, H0, NN, EMBD, NN, NN, EMBD, EMBD, 0, 1.f, nullptr, 0.f, 0.f, nullptr, 0.f);

  // 5) mask1[e] = ((S1 @ A0[e]) @ S1 != 0)
  for (int e = 0; e < 4; ++e) {
    fill(Bb[0], 0.f, NN2);
    k_adj<<<(NE_ + 255) / 256, 256, 0, stream>>>(Bb[0], ei[e], NE_, NN);
    gemm(Bb[1], S1, Bb[0], NN, NN, NN, NN, NN, NN, 0, 1.f, nullptr, 0.f, 0.f, nullptr, 0.f);
    gemm(Bb[2], Bb[1], S1, NN, NN, NN, NN, NN, NN, 0, 1.f, nullptr, 0.f, 0.f, nullptr, 0.f);
    k_thresh<<<eb, 256, 0, stream>>>(MK[e], Bb[2], NN2);
  }

  // 6) hetero_gnn #2 (dense, g2 weights, mask1) -> z1 in Z
  dense_layer(H1, EMBD, g2l1W, g2l1b);
  layernorm(Z, ACC, g2ln + 0, g2ln + HD);
  dense_layer(Z, HD, g2l2W, g2l2b);
  layernorm(Z, ACC, g2ln + 2 * HD, g2ln + 3 * HD);

  // 7) S2
  pairwise_S(Z, S2);

  // 8) h = S2 @ h -> H0
  gemm(H0, S2, H1, NN, EMBD, NN, NN, EMBD, EMBD, 0, 1.f, nullptr, 0.f, 0.f, nullptr, 0.f);

  // 9) mask2[e] = ((S2 @ mask1[e]) @ S2 != 0)  (in place on MK)
  for (int e = 0; e < 4; ++e) {
    gemm(Bb[1], S2, MK[e], NN, NN, NN, NN, NN, NN, 0, 1.f, nullptr, 0.f, 0.f, nullptr, 0.f);
    gemm(Bb[2], Bb[1], S2, NN, NN, NN, NN, NN, NN, 0, 1.f, nullptr, 0.f, 0.f, nullptr, 0.f);
    k_thresh<<<eb, 256, 0, stream>>>(MK[e], Bb[2], NN2);
  }

  // 10) hetero_gnn #3 (dense, g2 weights, mask2) -> z2 in Z
  dense_layer(H0, EMBD, g2l1W, g2l1b);
  layernorm(Z, ACC, g2ln + 0, g2ln + HD);
  dense_layer(Z, HD, g2l2W, g2l2b);
  layernorm(Z, ACC, g2ln + 2 * HD, g2ln + 3 * HD);

  // 11) global mean pool -> x2
  k_meanpool<<<1, 256, 0, stream>>>(Z, x2o, NN, HD);
  // 12) classifier + log_softmax
  k_classifier<<<1, 32, 0, stream>>>(x2o, clfW, clfb, lgo, HD, NCLS);
}
